// Qwen2MoeDecoderLayer_17136919511391
// MI455X (gfx1250) — matrix operand order repaired and checked by reading, not compiled, not run on hardware
//
#include <hip/hip_runtime.h>

typedef unsigned short u16;
typedef unsigned int   u32;
typedef unsigned long long u64;

typedef __attribute__((ext_vector_type(16))) __bf16 v16bf;
typedef __attribute__((ext_vector_type(8)))  float  v8f;

__device__ __forceinline__ u16 f2bf(float f) {
    u32 u = __builtin_bit_cast(u32, f);
    u32 r = u + 0x7fffu + ((u >> 16) & 1u);   // round-to-nearest-even
    return (u16)(r >> 16);
}
__device__ __forceinline__ float bf2f(u16 b) {
    u32 u = ((u32)b) << 16;
    return __builtin_bit_cast(float, u);
}

__device__ __forceinline__ void wait_async0() {
#if __has_builtin(__builtin_amdgcn_s_wait_asynccnt)
    __builtin_amdgcn_s_wait_asynccnt(0);
#else
    asm volatile("s_wait_asynccnt 0x0" ::: "memory");
#endif
}

// Async DMA: copy 16 bytes global -> LDS, tracked by ASYNCcnt
__device__ __forceinline__ void async_g2l_b128(u32 lds_off, const void* gptr) {
    asm volatile("global_load_async_to_lds_b128 %0, %1, off"
                 :: "v"(lds_off), "v"((u64)(uintptr_t)gptr) : "memory");
}

// ---------------------------------------------------------------------------
// WMMA GEMM:  C[M,N](f32) (+)= rowscale[m] * (A_bf16[M,K] @ B[K,N] * alpha + bias[n])
// Block tile 128x64, BK=32, 8 waves; wave owns 32x32 patch (4 WMMA accumulators).
// A tiles DMA'd global->LDS (async, bf16); B tiles via VGPRs (fp32->bf16),
// stored transposed.  Ping-pong LDS, one barrier per K-step.
// ---------------------------------------------------------------------------
template <bool B_BF16, bool HAS_BIAS, bool HAS_RS, bool ACCUM>
__global__ void __launch_bounds__(256)
gemm_wmma(const u16* __restrict__ A, const void* __restrict__ Bp,
          float* __restrict__ C, int K, int lda, int ldb, int ldc,
          const float* __restrict__ bias, float alpha,
          const float* __restrict__ rowscale, int rs_stride)
{
    __shared__ u16 As[2][128][40];
    __shared__ u16 Bs[2][64][40];

    const int tid  = threadIdx.x;
    const int lane = tid & 31;
    const int wid  = tid >> 5;
    const int bm   = blockIdx.y * 128;
    const int bn   = blockIdx.x * 64;
    const int wm   = (wid & 3) * 32;
    const int wn   = (wid >> 2) * 32;

    v8f c00 = {0.f,0.f,0.f,0.f,0.f,0.f,0.f,0.f};
    v8f c01 = c00, c10 = c00, c11 = c00;

    const int arow0 = tid >> 2,         ac0 = (tid & 3) * 8;
    const int arow1 = (tid + 256) >> 2, ac1 = ((tid + 256) & 3) * 8;
    const int bk = tid >> 3, bn0 = (tid & 7) * 8;

    auto issue_A = [&](int buf, int kt) {
        async_g2l_b128((u32)(uintptr_t)&As[buf][arow0][ac0],
                       A + (size_t)(bm + arow0) * lda + kt + ac0);
        async_g2l_b128((u32)(uintptr_t)&As[buf][arow1][ac1],
                       A + (size_t)(bm + arow1) * lda + kt + ac1);
    };

    float4 bf0, bf1; uint4 bq;
    auto load_B = [&](int kt) {
        if constexpr (B_BF16) {
            const u16* B = (const u16*)Bp;
            bq = *(const uint4*)(B + (size_t)(kt + bk) * ldb + bn + bn0);
        } else {
            const float* B = (const float*)Bp;
            const float* s = B + (size_t)(kt + bk) * ldb + bn + bn0;
            bf0 = *(const float4*)s;
            bf1 = *(const float4*)(s + 4);
        }
    };
    auto store_B = [&](int buf) {
        if constexpr (B_BF16) {
            const u16* ps = (const u16*)&bq;
            #pragma unroll
            for (int i = 0; i < 8; ++i) Bs[buf][bn0 + i][bk] = ps[i];
        } else {
            Bs[buf][bn0+0][bk] = f2bf(bf0.x); Bs[buf][bn0+1][bk] = f2bf(bf0.y);
            Bs[buf][bn0+2][bk] = f2bf(bf0.z); Bs[buf][bn0+3][bk] = f2bf(bf0.w);
            Bs[buf][bn0+4][bk] = f2bf(bf1.x); Bs[buf][bn0+5][bk] = f2bf(bf1.y);
            Bs[buf][bn0+6][bk] = f2bf(bf1.z); Bs[buf][bn0+7][bk] = f2bf(bf1.w);
        }
    };

    auto compute = [&](int buf) {
        const int hi = lane >> 4;
        const int r0 = wm + (lane & 15), r1 = r0 + 16;
        const int n0 = wn + (lane & 15), n1 = n0 + 16;
        union Frag { uint4 q[2]; v16bf v; } a0, a1, b0, b1;
        a0.q[0] = *(const uint4*)&As[buf][r0][hi * 8];
        a0.q[1] = *(const uint4*)&As[buf][r0][16 + hi * 8];
        a1.q[0] = *(const uint4*)&As[buf][r1][hi * 8];
        a1.q[1] = *(const uint4*)&As[buf][r1][16 + hi * 8];
        b0.q[0] = *(const uint4*)&Bs[buf][n0][hi * 8];
        b0.q[1] = *(const uint4*)&Bs[buf][n0][16 + hi * 8];
        b1.q[0] = *(const uint4*)&Bs[buf][n1][hi * 8];
        b1.q[1] = *(const uint4*)&Bs[buf][n1][16 + hi * 8];
        c00 = __builtin_amdgcn_wmma_f32_16x16x32_bf16(false, a0.v, false, b0.v, (short)0, c00, false, false);
        c01 = __builtin_amdgcn_wmma_f32_16x16x32_bf16(false, a0.v, false, b1.v, (short)0, c01, false, false);
        c10 = __builtin_amdgcn_wmma_f32_16x16x32_bf16(false, a1.v, false, b0.v, (short)0, c10, false, false);
        c11 = __builtin_amdgcn_wmma_f32_16x16x32_bf16(false, a1.v, false, b1.v, (short)0, c11, false, false);
    };

    const int nk = K >> 5;
    issue_A(0, 0);
    load_B(0);
    wait_async0();
    store_B(0);
    __syncthreads();

    for (int s = 0; s < nk; ++s) {
        const int cur = s & 1, nxt = cur ^ 1;
        const bool more = (s + 1) < nk;
        if (more) {
            issue_A(nxt, (s + 1) << 5);
            load_B((s + 1) << 5);
            if (((s + 2) << 5) < K) {
                if constexpr (B_BF16)
                    __builtin_prefetch((const u16*)Bp + (size_t)(((s + 2) << 5) + bk) * ldb + bn + bn0, 0, 1);
                else
                    __builtin_prefetch((const float*)Bp + (size_t)(((s + 2) << 5) + bk) * ldb + bn + bn0, 0, 1);
            }
        }
        compute(cur);
        if (more) {
            wait_async0();
            store_B(nxt);
            __syncthreads();
        }
    }

    const int hi = lane >> 4;
    #pragma unroll
    for (int j = 0; j < 8; ++j) {
        #pragma unroll
        for (int mi = 0; mi < 2; ++mi) {
            const int m = bm + wm + mi * 16 + hi * 8 + j;
            float rs = 1.0f;
            if constexpr (HAS_RS) rs = rowscale[(size_t)m * rs_stride];
            #pragma unroll
            for (int p = 0; p < 2; ++p) {
                const int n = bn + wn + p * 16 + (lane & 15);
                float v = (mi ? (p ? c11[j] : c10[j]) : (p ? c01[j] : c00[j])) * alpha;
                if constexpr (HAS_BIAS) v += bias[n];
                if constexpr (HAS_RS)   v *= rs;
                float* cp = C + (size_t)m * ldc + n;
                if constexpr (ACCUM) *cp += v; else *cp = v;
            }
        }
    }
}

// ---------------------------------------------------------------------------
__global__ void __launch_bounds__(256)
rmsnorm_bf16(const float* __restrict__ x, const float* __restrict__ w,
             u16* __restrict__ out, int Hdim)
{
    __shared__ float red[256];
    const int t = blockIdx.x;
    const float* xr = x + (size_t)t * Hdim;
    float ss = 0.f;
    for (int i = threadIdx.x; i < Hdim; i += 256) { float v = xr[i]; ss += v * v; }
    red[threadIdx.x] = ss; __syncthreads();
    for (int s = 128; s > 0; s >>= 1) {
        if (threadIdx.x < s) red[threadIdx.x] += red[threadIdx.x + s];
        __syncthreads();
    }
    const float rstd = rsqrtf(red[0] / (float)Hdim + 1e-6f);
    for (int i = threadIdx.x; i < Hdim; i += 256)
        out[(size_t)t * Hdim + i] = f2bf(xr[i] * rstd * w[i]);
}

__global__ void __launch_bounds__(256)
rope_pack(const float* __restrict__ q, const float* __restrict__ k,
          const float* __restrict__ v, const int* __restrict__ pos,
          u16* __restrict__ qo, u16* __restrict__ kto, u16* __restrict__ vo)
{
    const int t = blockIdx.x;
    const float p = (float)pos[t];
    for (int idx = threadIdx.x; idx < 2048; idx += 256) {
        int h = idx >> 7, d = idx & 127, i = d & 63;
        float inv = __powf(1000000.0f, -(float)i * (1.0f / 64.0f));
        float sn, cs; __sincosf(p * inv, &sn, &cs);
        float xv  = q[(size_t)t * 2048 + idx];
        float rot = (d < 64) ? -q[(size_t)t * 2048 + idx + 64]
                             :  q[(size_t)t * 2048 + idx - 64];
        qo[(size_t)h * 1024 * 128 + (size_t)t * 128 + d] = f2bf(xv * cs + rot * sn);
    }
    for (int idx = threadIdx.x; idx < 512; idx += 256) {
        int h = idx >> 7, d = idx & 127, i = d & 63;
        float inv = __powf(1000000.0f, -(float)i * (1.0f / 64.0f));
        float sn, cs; __sincosf(p * inv, &sn, &cs);
        float xv  = k[(size_t)t * 512 + idx];
        float rot = (d < 64) ? -k[(size_t)t * 512 + idx + 64]
                             :  k[(size_t)t * 512 + idx - 64];
        kto[(size_t)h * 128 * 1024 + (size_t)d * 1024 + t] = f2bf(xv * cs + rot * sn);
        vo[(size_t)h * 1024 * 128 + (size_t)t * 128 + d] = f2bf(v[(size_t)t * 512 + idx]);
    }
}

__global__ void __launch_bounds__(256)
softmax_causal(const float* __restrict__ scores, u16* __restrict__ probs, float scale)
{
    __shared__ float red[256];
    const int q = blockIdx.x, h = blockIdx.y;
    const float* row = scores + ((size_t)h * 1024 + q) * 1024;
    u16* prow = probs + ((size_t)h * 1024 + q) * 1024;
    const int n = q + 1;

    float mx = -3.4e38f;
    for (int k = threadIdx.x; k < n; k += 256) mx = fmaxf(mx, row[k] * scale);
    red[threadIdx.x] = mx; __syncthreads();
    for (int s = 128; s > 0; s >>= 1) {
        if (threadIdx.x < s) red[threadIdx.x] = fmaxf(red[threadIdx.x], red[threadIdx.x + s]);
        __syncthreads();
    }
    mx = red[0]; __syncthreads();

    float sum = 0.f;
    for (int k = threadIdx.x; k < n; k += 256) sum += __expf(row[k] * scale - mx);
    red[threadIdx.x] = sum; __syncthreads();
    for (int s = 128; s > 0; s >>= 1) {
        if (threadIdx.x < s) red[threadIdx.x] += red[threadIdx.x + s];
        __syncthreads();
    }
    const float inv = 1.0f / red[0];
    for (int k = threadIdx.x; k < 1024; k += 256)
        prow[k] = (k < n) ? f2bf(__expf(row[k] * scale - mx) * inv) : (u16)0;
}

__global__ void __launch_bounds__(256)
token_gates(const u16* __restrict__ xn, const float* __restrict__ rw,
            const float* __restrict__ sgw, float* __restrict__ weights,
            float* __restrict__ sg)
{
    __shared__ float red[256];
    __shared__ float logits[9];
    const int t = blockIdx.x;
    float loc[9];
    #pragma unroll
    for (int i = 0; i < 9; ++i) loc[i] = 0.f;
    for (int k = threadIdx.x; k < 2048; k += 256) {
        float xv = bf2f(xn[(size_t)t * 2048 + k]);
        #pragma unroll
        for (int e = 0; e < 8; ++e) loc[e] += xv * rw[(size_t)k * 8 + e];
        loc[8] += xv * sgw[k];
    }
    for (int i = 0; i < 9; ++i) {
        red[threadIdx.x] = loc[i]; __syncthreads();
        for (int s = 128; s > 0; s >>= 1) {
            if (threadIdx.x < s) red[threadIdx.x] += red[threadIdx.x + s];
            __syncthreads();
        }
        if (threadIdx.x == 0) logits[i] = red[0];
        __syncthreads();
    }
    if (threadIdx.x == 0) {
        float m = logits[0];
        for (int e = 1; e < 8; ++e) m = fmaxf(m, logits[e]);
        float p[8], sum = 0.f;
        for (int e = 0; e < 8; ++e) { p[e] = __expf(logits[e] - m); sum += p[e]; }
        for (int e = 0; e < 8; ++e) p[e] /= sum;
        int i1 = 0;
        for (int e = 1; e < 8; ++e) if (p[e] > p[i1]) i1 = e;
        int i2 = (i1 == 0) ? 1 : 0;
        for (int e = 0; e < 8; ++e) if (e != i1 && p[e] > p[i2]) i2 = e;
        for (int e = 0; e < 8; ++e)
            weights[(size_t)t * 8 + e] = (e == i1) ? p[i1] : ((e == i2) ? p[i2] : 0.f);
        sg[t] = 1.0f / (1.0f + __expf(-logits[8]));
    }
}

__global__ void __launch_bounds__(256)
silu_mul_bf16(const float* __restrict__ g, const float* __restrict__ u,
              u16* __restrict__ out, int count)
{
    int i = blockIdx.x * 256 + threadIdx.x;
    if (i < count) {
        float gv = g[i];
        out[i] = f2bf((gv / (1.0f + __expf(-gv))) * u[i]);
    }
}

__global__ void __launch_bounds__(256)
cvt_f32_bf16(const float* __restrict__ in, u16* __restrict__ out, int count)
{
    int i = blockIdx.x * 256 + threadIdx.x;
    if (i < count) out[i] = f2bf(in[i]);
}

// ---------------------------------------------------------------------------
extern "C" void kernel_launch(void* const* d_in, const int* in_sizes, int n_in,
                              void* d_out, int out_size, void* d_ws, size_t ws_size,
                              hipStream_t stream)
{
    (void)in_sizes; (void)n_in; (void)out_size; (void)ws_size;
    const int T = 1024, H = 2048, HD = 128, IM = 1408, IS = 5632;

    const float* x        = (const float*)d_in[0];
    const int*   pos      = (const int*)  d_in[1];
    const float* ln1      = (const float*)d_in[2];
    const float* ln2      = (const float*)d_in[3];
    const float* q_w      = (const float*)d_in[4];
    const float* q_b      = (const float*)d_in[5];
    const float* k_w      = (const float*)d_in[6];
    const float* k_b      = (const float*)d_in[7];
    const float* v_w      = (const float*)d_in[8];
    const float* v_b      = (const float*)d_in[9];
    const float* o_w      = (const float*)d_in[10];
    const float* router_w = (const float*)d_in[11];
    const float* e_gate   = (const float*)d_in[12];
    const float* e_up     = (const float*)d_in[13];
    const float* e_down   = (const float*)d_in[14];
    const float* s_gate   = (const float*)d_in[15];
    const float* s_up     = (const float*)d_in[16];
    const float* s_down   = (const float*)d_in[17];
    const float* sg_w     = (const float*)d_in[18];
    float* out = (float*)d_out;

    char* ws = (char*)d_ws;
    size_t off = 0;
    auto alloc = [&](size_t bytes) -> char* {
        char* p = ws + off;
        off = (off + bytes + 255) & ~(size_t)255;
        return p;
    };
    u16*   xn      = (u16*)  alloc((size_t)T * H * 2);
    float* qbuf    = (float*)alloc((size_t)T * H * 4);
    float* kbuf    = (float*)alloc((size_t)T * 512 * 4);
    float* vbuf    = (float*)alloc((size_t)T * 512 * 4);
    u16*   qbf     = (u16*)  alloc((size_t)T * H * 2);
    u16*   ktbf    = (u16*)  alloc((size_t)512 * T * 2);
    u16*   vbf     = (u16*)  alloc((size_t)512 * T * 2);
    float* h1      = (float*)alloc((size_t)T * H * 4);
    float* weights = (float*)alloc((size_t)T * 8 * 4);
    float* sgbuf   = (float*)alloc((size_t)T * 4);
    size_t scratch = off;
    float* scores  = (float*)alloc((size_t)16 * T * T * 4);
    u16*   probs   = (u16*)  alloc((size_t)16 * T * T * 2);
    float* attn    = (float*)alloc((size_t)T * H * 4);
    u16*   attnbf  = (u16*)  alloc((size_t)T * H * 2);
    off = scratch;
    float* gbuf    = (float*)alloc((size_t)T * IS * 4);
    float* ubuf    = (float*)alloc((size_t)T * IS * 4);
    u16*   actbf   = (u16*)  alloc((size_t)T * IS * 2);

    // ---- attention ----
    rmsnorm_bf16<<<T, 256, 0, stream>>>(x, ln1, xn, H);
    gemm_wmma<false,true,false,false><<<dim3(H/64,   T/128), 256, 0, stream>>>(
        xn, q_w, qbuf, H, H, H,   H,   q_b, 1.f, nullptr, 0);
    gemm_wmma<false,true,false,false><<<dim3(512/64, T/128), 256, 0, stream>>>(
        xn, k_w, kbuf, H, H, 512, 512, k_b, 1.f, nullptr, 0);
    gemm_wmma<false,true,false,false><<<dim3(512/64, T/128), 256, 0, stream>>>(
        xn, v_w, vbuf, H, H, 512, 512, v_b, 1.f, nullptr, 0);
    rope_pack<<<T, 256, 0, stream>>>(qbuf, kbuf, vbuf, pos, qbf, ktbf, vbf);

    for (int h = 0; h < 16; ++h)
        gemm_wmma<true,false,false,false><<<dim3(T/64, T/128), 256, 0, stream>>>(
            qbf + (size_t)h * T * HD, ktbf + (size_t)(h / 4) * HD * T,
            scores + (size_t)h * T * T, HD, HD, T, T, nullptr, 1.f, nullptr, 0);

    softmax_causal<<<dim3(T, 16), 256, 0, stream>>>(scores, probs, 0.08838834764831845f);

    for (int h = 0; h < 16; ++h)
        gemm_wmma<true,false,false,false><<<dim3(HD/64, T/128), 256, 0, stream>>>(
            probs + (size_t)h * T * T, vbf + (size_t)(h / 4) * T * HD,
            attn + (size_t)h * HD, T, T, HD, H, nullptr, 1.f, nullptr, 0);

    cvt_f32_bf16<<<(T * H) / 256, 256, 0, stream>>>(attn, attnbf, T * H);
    hipMemcpyAsync(h1, x, (size_t)T * H * 4, hipMemcpyDeviceToDevice, stream);
    gemm_wmma<false,false,false,true><<<dim3(H/64, T/128), 256, 0, stream>>>(
        attnbf, o_w, h1, H, H, H, H, nullptr, 1.f, nullptr, 0);   // h1 = x + attn @ o_w

    // ---- MoE ----
    rmsnorm_bf16<<<T, 256, 0, stream>>>(h1, ln2, xn, H);
    token_gates<<<T, 256, 0, stream>>>(xn, router_w, sg_w, weights, sgbuf);
    hipMemcpyAsync(out, h1, (size_t)T * H * 4, hipMemcpyDeviceToDevice, stream);

    for (int e = 0; e < 8; ++e) {
        gemm_wmma<false,false,false,false><<<dim3(IM/64, T/128), 256, 0, stream>>>(
            xn, e_gate + (size_t)e * H * IM, gbuf, H, H, IM, IM, nullptr, 1.f, nullptr, 0);
        gemm_wmma<false,false,false,false><<<dim3(IM/64, T/128), 256, 0, stream>>>(
            xn, e_up + (size_t)e * H * IM, ubuf, H, H, IM, IM, nullptr, 1.f, nullptr, 0);
        silu_mul_bf16<<<(T * IM) / 256, 256, 0, stream>>>(gbuf, ubuf, actbf, T * IM);
        gemm_wmma<false,false,true,true><<<dim3(H/64, T/128), 256, 0, stream>>>(
            actbf, e_down + (size_t)e * IM * H, out, IM, IM, H, H,
            nullptr, 1.f, weights + e, 8);
    }

    gemm_wmma<false,false,false,false><<<dim3(IS/64, T/128), 256, 0, stream>>>(
        xn, s_gate, gbuf, H, H, IS, IS, nullptr, 1.f, nullptr, 0);
    gemm_wmma<false,false,false,false><<<dim3(IS/64, T/128), 256, 0, stream>>>(
        xn, s_up, ubuf, H, H, IS, IS, nullptr, 1.f, nullptr, 0);
    silu_mul_bf16<<<(T * IS) / 256, 256, 0, stream>>>(gbuf, ubuf, actbf, T * IS);
    gemm_wmma<false,false,true,true><<<dim3(H/64, T/128), 256, 0, stream>>>(
        actbf, s_down, out, IS, IS, H, H, nullptr, 1.f, sgbuf, 1);
}
